// MultiHeadAttention_86492051407436
// MI455X (gfx1250) — matrix operand order repaired and checked by reading, not compiled, not run on hardware
//
#include <hip/hip_runtime.h>
#include <hip/hip_bf16.h>

// ---------------------------------------------------------------------------
// MHA forward for MI455X (gfx1250, wave32, WMMA f32_16x16x32_f16).
// B=2, T=2048, C=768, H=12, D=64.
// Pipeline: f32->f16 convert/transpose -> QKV GEMM (WMMA) -> flash attention
// (WMMA + online softmax, P staged via LDS) -> proj GEMM (WMMA) -> f32 out.
// ---------------------------------------------------------------------------

#define BATCH 2
#define TSEQ  2048
#define CDIM  768
#define NH    12
#define HD    64
#define BT    (BATCH * TSEQ)   // 4096 rows
#define N3C   (3 * CDIM)       // 2304

typedef __attribute__((ext_vector_type(16))) _Float16 v16h;
typedef __attribute__((ext_vector_type(8)))  _Float16 v8h;
typedef __attribute__((ext_vector_type(8)))  float    v8f;

// D = A(16x32) * B(32x16) + C, f16 in / f32 acc
__device__ __forceinline__ v8f wmma_f16(v16h a, v16h b, v8f c) {
    return __builtin_amdgcn_wmma_f32_16x16x32_f16(
        /*neg_a=*/false, a, /*neg_b=*/false, b,
        /*c_mod=*/(short)0, c, /*reuse_a=*/false, /*reuse_b=*/false);
}

// A-fragment (16x32 f16): lane holds row m0 + (lane&15); K elements at
// offsets [half*8, half*8+8) and [16+half*8, 16+half*8+8)  (half = lane>>4).
__device__ __forceinline__ v16h load_afrag(const _Float16* __restrict__ A,
                                           int lda, int m0, int kc, int lane) {
    const int nl = lane & 15, half = lane >> 4;
    const _Float16* p = A + (size_t)(m0 + nl) * lda + kc + half * 8;
    union { v16h v; v8h h[2]; } u;
    u.h[0] = *(const v8h*)(p);
    u.h[1] = *(const v8h*)(p + 16);
    return u.v;
}

// B-fragment (32x16 f16): lane holds COLUMN n0 + (lane&15); K elements at
// offsets [half*8, half*8+8) and [16+half*8, 16+half*8+8). Bt is the column-major view (i.e. B^T stored
// row-major with leading dim ldb).
__device__ __forceinline__ v16h load_bfrag(const _Float16* __restrict__ Bt,
                                           int ldb, int n0, int kc, int lane) {
    const int nl = lane & 15, half = lane >> 4;
    const _Float16* p = Bt + (size_t)(n0 + nl) * ldb + kc + half * 8;
    union { v16h v; v8h h[2]; } u;
    u.h[0] = *(const v8h*)(p);
    u.h[1] = *(const v8h*)(p + 16);
    return u.v;
}

// ---------------------------------------------------------------------------
// f32 -> f16 elementwise convert
__global__ void cvt_f32_f16(const float* __restrict__ src,
                            _Float16* __restrict__ dst, int n) {
    int i = blockIdx.x * blockDim.x + threadIdx.x;
    if (i < n) dst[i] = (_Float16)src[i];
}

// W [rows, cols] f32 -> Wt [cols, rows] f16
__global__ void transpose_f32_f16(const float* __restrict__ W,
                                  _Float16* __restrict__ Wt,
                                  int rows, int cols) {
    int i = blockIdx.x * blockDim.x + threadIdx.x;
    if (i < rows * cols) {
        int r = i / cols, c = i - r * cols;
        Wt[(size_t)c * rows + r] = (_Float16)W[i];
    }
}

// ---------------------------------------------------------------------------
// QKV GEMM: qkv[m,n] = sum_k Xh[m,k] * Wt[n,k] + bias[n]; scatter into
// Q [B,H,T,D] (scaled by 1/8), K [B,H,T,D], Vt [B,H,D,T], all f16.
// One wave per 32x64 output tile. 4 waves / block. 128*36 = 4608 tiles exact.
__global__ __launch_bounds__(128) void qkv_gemm_kernel(
    const _Float16* __restrict__ Xh, const _Float16* __restrict__ Wt,
    const float* __restrict__ bias,
    _Float16* __restrict__ Qh, _Float16* __restrict__ Kh,
    _Float16* __restrict__ Vt) {
    const int lane = threadIdx.x & 31;
    const int wid  = blockIdx.x * 4 + (threadIdx.x >> 5);
    const int MT   = BT / 32;                 // 128
    const int mt   = wid % MT, nt = wid / MT; // nt in [0,36)
    const int m0   = mt * 32, n0 = nt * 64;

    v8f acc[2][4] = {};
    for (int kc = 0; kc < CDIM; kc += 32) {
        v16h a0 = load_afrag(Xh, CDIM, m0,      kc, lane);
        v16h a1 = load_afrag(Xh, CDIM, m0 + 16, kc, lane);
#pragma unroll
        for (int j = 0; j < 4; ++j) {
            v16h b = load_bfrag(Wt, CDIM, n0 + j * 16, kc, lane);
            acc[0][j] = wmma_f16(a0, b, acc[0][j]);
            acc[1][j] = wmma_f16(a1, b, acc[1][j]);
        }
    }

    const int nl = lane & 15, half = lane >> 4;
#pragma unroll
    for (int j = 0; j < 4; ++j) {
        int n   = n0 + j * 16 + nl;
        int sec = n / CDIM;            // 0=q, 1=k, 2=v (uniform per 64-tile)
        int c   = n - sec * CDIM;
        int h   = c >> 6, d = c & 63;
        float bv = bias[n];
#pragma unroll
        for (int i = 0; i < 2; ++i) {
#pragma unroll
            for (int r = 0; r < 8; ++r) {
                int m  = m0 + i * 16 + r + 8 * half;
                int bi = m >> 11;          // m / TSEQ
                int t  = m & (TSEQ - 1);
                float val = acc[i][j][r] + bv;
                size_t head = (size_t)(bi * NH + h);
                if (sec == 0)
                    Qh[(head * TSEQ + t) * HD + d] = (_Float16)(val * 0.125f);
                else if (sec == 1)
                    Kh[(head * TSEQ + t) * HD + d] = (_Float16)val;
                else
                    Vt[(head * HD + d) * TSEQ + t] = (_Float16)val;
            }
        }
    }
}

// ---------------------------------------------------------------------------
// Flash attention: one wave per (b,h,16-query tile). 32-key blocks.
// S = Q K^T (4 WMMA), online softmax (shfl row reductions), P via LDS,
// O += P V (4 WMMA). Oh written as [B*T, C] f16.
__global__ __launch_bounds__(128) void attn_kernel(
    const _Float16* __restrict__ Qh, const _Float16* __restrict__ Kh,
    const _Float16* __restrict__ Vt, _Float16* __restrict__ Oh) {
    __shared__ _Float16 plds[4][16 * 32];   // per-wave 16x32 P tile (4 KB)

    const int lane = threadIdx.x & 31;
    const int wave = threadIdx.x >> 5;
    const int wid  = blockIdx.x * 4 + wave;       // [0, 3072)
    const int qt   = wid & 127;                   // 128 query tiles
    const int bh   = wid >> 7;                    // [0, 24)
    const int nl   = lane & 15, half = lane >> 4;

    const _Float16* Qb = Qh + (size_t)bh * TSEQ * HD;
    const _Float16* Kb = Kh + (size_t)bh * TSEQ * HD;
    const _Float16* Vb = Vt + (size_t)bh * HD * TSEQ;

    const int q0 = qt * 16;

    // Q A-fragments (pre-scaled by 1/sqrt(D) in kernel 1)
    v16h qa0 = load_afrag(Qb, HD, q0, 0,  lane);
    v16h qa1 = load_afrag(Qb, HD, q0, 32, lane);

    v8f o[4] = {};
    float mrow[8], lrow[8];
#pragma unroll
    for (int r = 0; r < 8; ++r) { mrow[r] = -1e30f; lrow[r] = 0.0f; }

    const int kend = q0 + 16;   // keys < kend may be unmasked
    for (int kb = 0; kb < kend; kb += 32) {
        // ---- logits for 32 keys ----
        v8f s0 = {}, s1 = {};
        {
            v16h b00 = load_bfrag(Kb, HD, kb, 0,  lane);
            v16h b01 = load_bfrag(Kb, HD, kb, 32, lane);
            s0 = wmma_f16(qa0, b00, s0);
            s0 = wmma_f16(qa1, b01, s0);
        }
        const bool tile1 = (kb + 16) < kend;      // wave-uniform
        if (tile1) {
            v16h b10 = load_bfrag(Kb, HD, kb + 16, 0,  lane);
            v16h b11 = load_bfrag(Kb, HD, kb + 16, 32, lane);
            s1 = wmma_f16(qa0, b10, s1);
            s1 = wmma_f16(qa1, b11, s1);
        }

        // ---- causal mask + online softmax ----
        const int key0 = kb + nl;
        const int key1 = kb + 16 + nl;
        _Float16* lp = plds[wave];
#pragma unroll
        for (int r = 0; r < 8; ++r) {
            int q = q0 + r + 8 * half;
            float v0 = (key0 <= q) ? s0[r] : -1e30f;
            float v1 = (tile1 && key1 <= q) ? s1[r] : -1e30f;
            float mb = fmaxf(v0, v1);
#pragma unroll
            for (int off = 8; off > 0; off >>= 1)
                mb = fmaxf(mb, __shfl_xor(mb, off, 16));
            float mnew  = fmaxf(mrow[r], mb);
            float alpha = __expf(mrow[r] - mnew);
            float p0 = __expf(v0 - mnew);
            float p1 = __expf(v1 - mnew);
            float rs = p0 + p1;
#pragma unroll
            for (int off = 8; off > 0; off >>= 1)
                rs += __shfl_xor(rs, off, 16);
            lrow[r] = lrow[r] * alpha + rs;
            mrow[r] = mnew;
            o[0][r] *= alpha; o[1][r] *= alpha;
            o[2][r] *= alpha; o[3][r] *= alpha;
            // store P (C/D layout -> row-major LDS tile)
            int row = r + 8 * half;
            lp[row * 32 + nl]      = (_Float16)p0;
            lp[row * 32 + 16 + nl] = (_Float16)p1;
        }
        __asm__ volatile("s_wait_dscnt 0" ::: "memory");

        // reload P as A-fragment (16x32)
        v16h pa;
        {
            const _Float16* pr = lp + nl * 32 + half * 8;
            union { v16h v; v8h h[2]; } u;
            u.h[0] = *(const v8h*)(pr);
            u.h[1] = *(const v8h*)(pr + 16);
            pa = u.v;
        }
        // ---- O += P * V ---- (Vt rows are columns of V)
#pragma unroll
        for (int j = 0; j < 4; ++j) {
            v16h bv = load_bfrag(Vb, TSEQ, j * 16, kb, lane);
            o[j] = wmma_f16(pa, bv, o[j]);
        }
    }

    // ---- normalize + write Oh [B*T, C] f16 ----
    float invl[8];
#pragma unroll
    for (int r = 0; r < 8; ++r) invl[r] = 1.0f / lrow[r];
    const int b = bh / NH, h = bh - b * NH;
#pragma unroll
    for (int j = 0; j < 4; ++j) {
#pragma unroll
        for (int r = 0; r < 8; ++r) {
            int q   = q0 + r + 8 * half;
            int col = h * HD + j * 16 + nl;
            Oh[(size_t)(b * TSEQ + q) * CDIM + col] = (_Float16)(o[j][r] * invl[r]);
        }
    }
}

// ---------------------------------------------------------------------------
// Output projection: out[m,n] = sum_k Oh[m,k] * WpT[n,k] + bias[n], f32 out.
// 128*12 = 1536 tiles exact.
__global__ __launch_bounds__(128) void proj_gemm_kernel(
    const _Float16* __restrict__ Oh, const _Float16* __restrict__ WpT,
    const float* __restrict__ bias, float* __restrict__ out) {
    const int lane = threadIdx.x & 31;
    const int wid  = blockIdx.x * 4 + (threadIdx.x >> 5);
    const int MT   = BT / 32;                  // 128
    const int mt   = wid % MT, nt = wid / MT;  // nt in [0,12)
    const int m0   = mt * 32, n0 = nt * 64;

    v8f acc[2][4] = {};
    for (int kc = 0; kc < CDIM; kc += 32) {
        v16h a0 = load_afrag(Oh, CDIM, m0,      kc, lane);
        v16h a1 = load_afrag(Oh, CDIM, m0 + 16, kc, lane);
#pragma unroll
        for (int j = 0; j < 4; ++j) {
            v16h b = load_bfrag(WpT, CDIM, n0 + j * 16, kc, lane);
            acc[0][j] = wmma_f16(a0, b, acc[0][j]);
            acc[1][j] = wmma_f16(a1, b, acc[1][j]);
        }
    }

    const int nl = lane & 15, half = lane >> 4;
#pragma unroll
    for (int j = 0; j < 4; ++j) {
        int n = n0 + j * 16 + nl;
        float bv = bias[n];
#pragma unroll
        for (int i = 0; i < 2; ++i) {
#pragma unroll
            for (int r = 0; r < 8; ++r) {
                int m = m0 + i * 16 + r + 8 * half;
                out[(size_t)m * CDIM + n] = acc[i][j][r] + bv;
            }
        }
    }
}

// ---------------------------------------------------------------------------
extern "C" void kernel_launch(void* const* d_in, const int* in_sizes, int n_in,
                              void* d_out, int out_size, void* d_ws, size_t ws_size,
                              hipStream_t stream) {
    (void)in_sizes; (void)n_in; (void)out_size; (void)ws_size;
    const float* x     = (const float*)d_in[0];   // [B,T,C]
    const float* Wqkv  = (const float*)d_in[1];   // [C,3C]
    const float* bqkv  = (const float*)d_in[2];   // [3C]
    const float* Wproj = (const float*)d_in[3];   // [C,C]
    const float* bproj = (const float*)d_in[4];   // [C]
    float* out = (float*)d_out;                   // [B,T,C] f32

    // workspace carve (all region sizes are multiples of 16 bytes)
    char* p = (char*)d_ws;
    _Float16* Xh    = (_Float16*)p; p += (size_t)BT * CDIM   * 2;  // 6 MB
    _Float16* WqT   = (_Float16*)p; p += (size_t)N3C * CDIM  * 2;  // 3.375 MB
    _Float16* WpT   = (_Float16*)p; p += (size_t)CDIM * CDIM * 2;  // 1.125 MB
    _Float16* Qh    = (_Float16*)p; p += (size_t)BATCH * NH * TSEQ * HD * 2;
    _Float16* Kh    = (_Float16*)p; p += (size_t)BATCH * NH * TSEQ * HD * 2;
    _Float16* Vt    = (_Float16*)p; p += (size_t)BATCH * NH * HD * TSEQ * 2;
    _Float16* Ohbuf = (_Float16*)p; p += (size_t)BT * CDIM * 2;

    const int nx = BT * CDIM;
    cvt_f32_f16<<<(nx + 255) / 256, 256, 0, stream>>>(x, Xh, nx);
    transpose_f32_f16<<<(CDIM * N3C + 255) / 256, 256, 0, stream>>>(Wqkv, WqT, CDIM, N3C);
    transpose_f32_f16<<<(CDIM * CDIM + 255) / 256, 256, 0, stream>>>(Wproj, WpT, CDIM, CDIM);

    // 128*36 = 4608 waves / 4 per block
    qkv_gemm_kernel<<<1152, 128, 0, stream>>>(Xh, WqT, bqkv, Qh, Kh, Vt);
    // 2*12*128 = 3072 waves / 4 per block
    attn_kernel<<<768, 128, 0, stream>>>(Qh, Kh, Vt, Ohbuf);
    // 128*12 = 1536 waves / 4 per block
    proj_gemm_kernel<<<384, 128, 0, stream>>>(Ohbuf, WpT, bproj, out);
}
